// EdgeConv_90503550861384
// MI455X (gfx1250) — matrix operand order repaired and checked by reading, not compiled, not run on hardware
//
#include <hip/hip_runtime.h>

#define BB 4
#define CC 64
#define NN 4096
#define KK 20
#define OO 64

typedef __attribute__((ext_vector_type(16))) __bf16 v16bf;
typedef __attribute__((ext_vector_type(8)))  float  v8f;

union ABf {
  uint4 q[2];   // 32 bytes = 16 bf16, element j lives in VGPR j>>1 (matches ISA striping)
  v16bf v;
};

__device__ __forceinline__ unsigned short f2bf(float f) {
  unsigned u = __float_as_uint(f);
  unsigned r = (u + 0x7FFFu + ((u >> 16) & 1u)) >> 16;   // round-to-nearest-even
  return (unsigned short)r;
}

// ---------------------------------------------------------------------------
// 1) points (B,C,N) f32 -> Xb (B,N,C) bf16 ; squared norms sqn (B,N) f32
// ---------------------------------------------------------------------------
__global__ void prep_kernel(const float* __restrict__ points,
                            unsigned short* __restrict__ Xb,
                            float* __restrict__ sqn) {
  int id = blockIdx.x * blockDim.x + threadIdx.x;  // B*N threads
  int b = id / NN, n = id % NN;
  const float* p = points + (size_t)b * CC * NN + n;   // stride N over c: coalesced across threads
  unsigned short* xrow = Xb + (size_t)id * CC;
  float s = 0.f;
#pragma unroll 8
  for (int c = 0; c < CC; ++c) {
    float v = p[(size_t)c * NN];
    s += v * v;
    xrow[c] = f2bf(v);
  }
  sqn[id] = s;
}

// ---------------------------------------------------------------------------
// 2) Wcat[o2][c] bf16: o2<64 -> W1 ; o2>=64 -> (W2 - W1)
// ---------------------------------------------------------------------------
__global__ void wprep_kernel(const float* __restrict__ W,
                             unsigned short* __restrict__ Wcat) {
  int id = blockIdx.x * blockDim.x + threadIdx.x;  // 128*64 threads
  int o2 = id >> 6, c = id & 63;
  float v;
  if (o2 < OO) v = W[o2 * 2 * CC + c];
  else { int o = o2 - OO; v = W[o * 2 * CC + CC + c] - W[o * 2 * CC + c]; }
  Wcat[id] = f2bf(v);
}

// ---------------------------------------------------------------------------
// 3) QP[b][n][o2] = X[b][n][:] . Wcat[o2][:]   (o2 in [0,128))  via WMMA bf16
//    block = 256 thr (8 waves); block does 16 rows x 128 outs; wave w -> out tile w
// ---------------------------------------------------------------------------
__global__ void __launch_bounds__(256) qp_wmma_kernel(
    const unsigned short* __restrict__ Xb,
    const unsigned short* __restrict__ Wcat,
    float* __restrict__ QP) {
  int b = blockIdx.y;
  int rowBase = blockIdx.x * 16;
  int wave = threadIdx.x >> 5;
  int lane = threadIdx.x & 31;
  int h = lane >> 4, mn = lane & 15;

  const unsigned short* xrow = Xb + (size_t)(b * NN + rowBase + mn) * CC;
  const unsigned short* wrow = Wcat + (size_t)(wave * 16 + mn) * CC;

  v8f acc = {};
#pragma unroll
  for (int kt = 0; kt < CC; kt += 32) {
    ABf A, Bm;
    // A (16x32 bf16): lane(h,m): K = kt+8h+{0..7} then kt+16+8h+{0..7}
    A.q[0] = *(const uint4*)(xrow + kt + 8 * h);
    A.q[1] = *(const uint4*)(xrow + kt + 16 + 8 * h);
    // B (32x16 bf16): lane(h,n): K = kt+8h+{0..7} then kt+16+8h+{0..7} from Wcat row
    Bm.q[0] = *(const uint4*)(wrow + kt + 8 * h);
    Bm.q[1] = *(const uint4*)(wrow + kt + 16 + 8 * h);
    acc = __builtin_amdgcn_wmma_f32_16x16x32_bf16(false, A.v, false, Bm.v,
                                                  (short)0, acc, false, false);
  }
  // D layout: lane(h,nn), VGPR r -> [M = 8h+r][N = nn]
  float* dst = QP + (size_t)(b * NN + rowBase + 8 * h) * 128 + wave * 16 + mn;
#pragma unroll
  for (int r = 0; r < 8; ++r) dst[(size_t)r * 128] = acc[r];
}

// ---------------------------------------------------------------------------
// 4) Fused pairwise-distance (WMMA bf16, f32 accum) + streaming top-20 per row.
//    Block: 64 rows x all 4096 cols, 64-col chunks through LDS.
//    Wave w owns sub-tiles 2w,2w+1 -> both share A row-tile mt = w>>1, so the
//    A fragments are hoisted out of the column loop (held in 16 VGPRs).
//    Ranking key per row i: 2*inner - sq[j]  (the -sq[i] term is row-constant).
// ---------------------------------------------------------------------------
__global__ void __launch_bounds__(256) dist_topk_kernel(
    const unsigned short* __restrict__ Xb,
    const float* __restrict__ sqn,
    int* __restrict__ idxout) {
  __shared__ float tileT[64 * 68];   // [col][row], pad 68 to dodge bank conflicts
  __shared__ float scol[64];

  int b = blockIdx.y;
  int rowBase = blockIdx.x * 64;
  int tid = threadIdx.x;
  int wave = tid >> 5, lane = tid & 31;
  int h = lane >> 4, mn = lane & 15;
  int t0 = wave * 2;                 // 16 sub-tiles (4x4), 2 per wave
  int mt = wave >> 1;                // shared A row-tile for both sub-tiles

  // A fragments: invariant over the whole column sweep
  const unsigned short* arow = Xb + (size_t)(b * NN + rowBase + mt * 16 + mn) * CC;
  ABf A0, A1;
  A0.q[0] = *(const uint4*)(arow + 8 * h);
  A0.q[1] = *(const uint4*)(arow + 16 + 8 * h);
  A1.q[0] = *(const uint4*)(arow + 32 + 8 * h);
  A1.q[1] = *(const uint4*)(arow + 48 + 8 * h);

  float kv[KK]; int ki[KK];          // sorted descending; only threads<64 use them
#pragma unroll
  for (int j = 0; j < KK; ++j) { kv[j] = -3.4e38f; ki[j] = 0; }

  const float* sq_b = sqn + (size_t)b * NN;

  for (int cb = 0; cb < NN; cb += 64) {
    v8f acc[2] = {};
#pragma unroll
    for (int u = 0; u < 2; ++u) {
      int nt = (t0 + u) & 3;
      // inner = X.X^T: B-matrix columns are X rows -> contiguous loads, no transpose
      const unsigned short* brow = Xb + (size_t)(b * NN + cb + nt * 16 + mn) * CC;
      ABf Bm;
      Bm.q[0] = *(const uint4*)(brow + 8 * h);
      Bm.q[1] = *(const uint4*)(brow + 16 + 8 * h);
      acc[u] = __builtin_amdgcn_wmma_f32_16x16x32_bf16(false, A0.v, false, Bm.v,
                                                       (short)0, acc[u], false, false);
      Bm.q[0] = *(const uint4*)(brow + 32 + 8 * h);
      Bm.q[1] = *(const uint4*)(brow + 48 + 8 * h);
      acc[u] = __builtin_amdgcn_wmma_f32_16x16x32_bf16(false, A1.v, false, Bm.v,
                                                       (short)0, acc[u], false, false);
    }
    __syncthreads();   // previous chunk's scan finished before overwriting LDS
#pragma unroll
    for (int u = 0; u < 2; ++u) {
      int nt = (t0 + u) & 3;
      float* dst = tileT + (nt * 16 + mn) * 68 + mt * 16 + 8 * h;
      *(float4*)(dst)     = make_float4(acc[u][0], acc[u][1], acc[u][2], acc[u][3]);
      *(float4*)(dst + 4) = make_float4(acc[u][4], acc[u][5], acc[u][6], acc[u][7]);
    }
    if (tid >= 64 && tid < 128) scol[tid - 64] = sq_b[cb + (tid - 64)];
    __syncthreads();
    if (tid < 64) {                  // thread r scans its row across 64 cols
#pragma unroll 4
      for (int j = 0; j < 64; ++j) {
        float key = 2.0f * tileT[j * 68 + tid] - scol[j];
        if (key > kv[KK - 1]) {      // fully-unrolled reg-only sorted insert
          float ckv = key; int cki = cb + j;
#pragma unroll
          for (int s = 0; s < KK; ++s) {
            bool gt = ckv > kv[s];
            float tv = kv[s]; int ti = ki[s];
            kv[s] = gt ? ckv : tv; ki[s] = gt ? cki : ti;
            ckv = gt ? tv : ckv;     cki = gt ? ti : cki;
          }
        }
      }
    }
  }
  if (tid < 64) {
    int* dst = idxout + (size_t)(b * NN + rowBase + tid) * KK;
#pragma unroll
    for (int j = 0; j < KK; ++j) dst[j] = ki[j];
  }
}

// ---------------------------------------------------------------------------
// 5) Gather stats: y = Q[idx] + P. Accumulate sum/sumsq per (b,o,group) and
//    per-point max (max over k commutes with monotone norm+LeakyReLU).
// ---------------------------------------------------------------------------
__global__ void stats_kernel(const float* __restrict__ QP,
                             const int* __restrict__ idxbuf,
                             float* __restrict__ mbuf,
                             float* __restrict__ psum,
                             float* __restrict__ psq) {
  __shared__ int sidx[64 * KK];
  int b = blockIdx.y;
  int g = blockIdx.x;          // 64 groups of 64 points
  int nbase = g * 64;
  int o = threadIdx.x;         // 64 threads = output channels
  for (int i = o; i < 64 * KK; i += 64)
    sidx[i] = idxbuf[(size_t)(b * NN + nbase) * KK + i];
  __syncthreads();

  const float* QPb = QP + (size_t)b * NN * 128;
  float sum = 0.f, sq = 0.f;
  for (int nn = 0; nn < 64; ++nn) {
    int n = nbase + nn;
    float P = QPb[(size_t)n * 128 + 64 + o];
    float mx = -3.4e38f;
#pragma unroll 4
    for (int k = 0; k < KK; ++k) {
      int j = sidx[nn * KK + k];
      float q = QPb[(size_t)j * 128 + o];   // 64 consecutive o -> coalesced 256B
      float y = q + P;
      sum += y; sq += y * y;
      mx = fmaxf(mx, q);
    }
    mbuf[(size_t)(b * OO + o) * NN + n] = mx + P;
  }
  int ch = b * OO + o;
  psum[ch * 64 + g] = sum;
  psq[ch * 64 + g]  = sq;
}

// ---------------------------------------------------------------------------
// 6) Deterministic reduction -> mean, rsqrt(var+eps) per (b,o)
// ---------------------------------------------------------------------------
__global__ void finalize_stats_kernel(const float* __restrict__ psum,
                                      const float* __restrict__ psq,
                                      float* __restrict__ ms) {
  int ch = threadIdx.x;        // 256 channels total
  float s = 0.f, ss = 0.f;
  for (int g = 0; g < 64; ++g) { s += psum[ch * 64 + g]; ss += psq[ch * 64 + g]; }
  const float inv = 1.0f / (float)(NN * KK);
  float mean = s * inv;
  float var  = ss * inv - mean * mean;
  ms[ch * 2]     = mean;
  ms[ch * 2 + 1] = rsqrtf(var + 1e-5f);
}

// ---------------------------------------------------------------------------
// 7) out[b][o][n] = LeakyReLU((m - mean) * invstd)
// ---------------------------------------------------------------------------
__global__ void final_kernel(const float* __restrict__ mbuf,
                             const float* __restrict__ ms,
                             float* __restrict__ out) {
  int id = blockIdx.x * blockDim.x + threadIdx.x;  // B*OO*NN
  int ch = id >> 12;                               // / NN
  float v = (mbuf[id] - ms[ch * 2]) * ms[ch * 2 + 1];
  out[id] = v >= 0.f ? v : 0.2f * v;
}

// ---------------------------------------------------------------------------
extern "C" void kernel_launch(void* const* d_in, const int* in_sizes, int n_in,
                              void* d_out, int out_size, void* d_ws, size_t ws_size,
                              hipStream_t stream) {
  (void)in_sizes; (void)n_in; (void)out_size; (void)ws_size;
  const float* points = (const float*)d_in[0];   // (B,C,N) f32
  const float* W      = (const float*)d_in[1];   // (OUT,2C) f32
  float* out = (float*)d_out;                    // (B,OUT,N) f32

  char* ws = (char*)d_ws;
  size_t off = 0;
  auto alloc = [&](size_t bytes) -> void* {
    void* p = ws + off;
    off = (off + bytes + 255) & ~(size_t)255;
    return p;
  };
  unsigned short* Xb   = (unsigned short*)alloc((size_t)BB * NN * CC * 2);   // 2 MB
  float* sqn           = (float*)alloc((size_t)BB * NN * 4);                 // 64 KB
  unsigned short* Wcat = (unsigned short*)alloc((size_t)128 * CC * 2);       // 16 KB
  float* QP            = (float*)alloc((size_t)BB * NN * 128 * 4);           // 8 MB
  int*   idxbuf        = (int*)alloc((size_t)BB * NN * KK * 4);              // 1.25 MB
  float* mbuf          = (float*)alloc((size_t)BB * OO * NN * 4);            // 4 MB
  float* psum          = (float*)alloc((size_t)256 * 64 * 4);
  float* psq           = (float*)alloc((size_t)256 * 64 * 4);
  float* ms            = (float*)alloc((size_t)256 * 2 * 4);

  prep_kernel<<<BB * NN / 256, 256, 0, stream>>>(points, Xb, sqn);
  wprep_kernel<<<128 * CC / 256, 256, 0, stream>>>(W, Wcat);
  qp_wmma_kernel<<<dim3(NN / 16, BB), 256, 0, stream>>>(Xb, Wcat, QP);
  dist_topk_kernel<<<dim3(NN / 64, BB), 256, 0, stream>>>(Xb, sqn, idxbuf);
  stats_kernel<<<dim3(64, BB), 64, 0, stream>>>(QP, idxbuf, mbuf, psum, psq);
  finalize_stats_kernel<<<1, 256, 0, stream>>>(psum, psq, ms);
  final_kernel<<<BB * OO * NN / 256, 256, 0, stream>>>(mbuf, ms, out);
}
